// QuantumCoherentLayer_23639499997373
// MI455X (gfx1250) — matrix operand order repaired and checked by reading, not compiled, not run on hardware
//
#include <hip/hip_runtime.h>

typedef __attribute__((ext_vector_type(16))) _Float16 v16h;
typedef __attribute__((ext_vector_type(8)))  _Float16 v8h;
typedef __attribute__((ext_vector_type(8)))  float    v8f;

// ---------------------------------------------------------------------------
// Kernel 0: scalar prep.  alpha, column sums of softmax(cos(phase*J_sym)),
// fused output constants.  sc[0]=alpha^2/64, sc[1]=alpha(1-alpha)/64,
// sc[2..9]=w[j].
// ---------------------------------------------------------------------------
__global__ void qc_scalars_kernel(const float* __restrict__ J,
                                  const float* __restrict__ t,
                                  float* __restrict__ sc) {
    if (threadIdx.x == 0 && blockIdx.x == 0) {
        const float tv = t[0];
        float alpha = expf(-tv / 51.0f);
        alpha = fminf(fmaxf(alpha, 0.0f), 1.0f);
        const float phase = 2.0f * 3.14159265358979323846f * 20.0f * tv / 1000.0f;
        float wsum[8];
        #pragma unroll
        for (int j = 0; j < 8; ++j) wsum[j] = 0.0f;
        for (int i = 0; i < 8; ++i) {
            float row[8];
            float mx = -3.0e38f;
            for (int j = 0; j < 8; ++j) {
                const float js = 0.5f * (J[i * 8 + j] + J[j * 8 + i]);
                row[j] = cosf(phase * js);
                mx = fmaxf(mx, row[j]);
            }
            float s = 0.0f;
            for (int j = 0; j < 8; ++j) { row[j] = expf(row[j] - mx); s += row[j]; }
            const float inv = 1.0f / s;
            for (int j = 0; j < 8; ++j) wsum[j] += row[j] * inv;
        }
        sc[0] = alpha * alpha / 64.0f;
        sc[1] = alpha * (1.0f - alpha) / 64.0f;
        #pragma unroll
        for (int j = 0; j < 8; ++j) sc[2 + j] = wsum[j];
    }
}

// ---------------------------------------------------------------------------
// Kernel 1: W_eff[k,h] = sum_j w[j] * W_paths[j,k,h]  -> f16
// ---------------------------------------------------------------------------
__global__ void qc_weff_kernel(const float* __restrict__ Wp,
                               const float* __restrict__ sc,
                               _Float16* __restrict__ weff) {
    const int idx = blockIdx.x * 256 + threadIdx.x;   // 0 .. 1024*1024-1
    float acc = 0.0f;
    #pragma unroll
    for (int j = 0; j < 8; ++j)
        acc += sc[2 + j] * Wp[j * 1048576 + idx];
    weff[idx] = (_Float16)acc;
}

// ---------------------------------------------------------------------------
// WMMA GEMM: C[2048x1024] = A[2048x1024] * B[1024x1024]
// Block tile 128x128, K-step 32, 8 wave32 per block; wave w owns rows
// [16w,16w+16) of the tile -> 8 v_wmma_f32_16x16x32_f16 per K-step.
// HALF_IN: inputs already f16 (second GEMM); otherwise convert f32->f16
// while staging into LDS.
// FINAL:   fuse coherent-state epilogue and write f32 output; else write
// f16 "base" for the next GEMM.
// ---------------------------------------------------------------------------
template <bool HALF_IN, bool FINAL>
__global__ __launch_bounds__(256) void qc_gemm_kernel(
    const void* __restrict__ Av, const void* __restrict__ Bv,
    void* __restrict__ outv,
    const float* __restrict__ coh, const float* __restrict__ sc) {

    constexpr int N = 1024, K = 1024;
    constexpr int BM = 128, BN = 128, BK = 32;
    constexpr int BSTR = 40;                      // halves per LDS B row (16B aligned)

    __shared__ _Float16 a_lds[BM * BK];           // [m][k], stride 32 halves
    __shared__ _Float16 b_lds[BN * BSTR];         // [n][k], stride 40 halves

    const int tid  = threadIdx.x;
    const int wv   = tid >> 5;
    const int lane = tid & 31;
    const int bm   = blockIdx.y;
    const int bn   = blockIdx.x;

    const float*    Af = (const float*)Av;
    const _Float16* Ah = (const _Float16*)Av;
    const float*    Bf = (const float*)Bv;
    const _Float16* Bh = (const _Float16*)Bv;

    v8f acc[8];
    #pragma unroll
    for (int i = 0; i < 8; ++i) acc[i] = (v8f){0.f,0.f,0.f,0.f,0.f,0.f,0.f,0.f};

    const int mrow = wv * 16 + (lane & 15);       // local A row for this lane
    const int kb   = (lane >> 4) * 8;             // A fragment K base (0 or 8)
    const int kkb  = (lane >> 4) * 8;             // B fragment K base (0 or 8)

    for (int k0 = 0; k0 < K; k0 += BK) {
        // ---- stage A tile: 128x32 (row-major) ----
        #pragma unroll
        for (int i = 0; i < 16; ++i) {
            const int idx = tid + i * 256;
            const int r = idx >> 5, c = idx & 31;
            const long g = (long)(bm * BM + r) * K + (k0 + c);
            a_lds[idx] = HALF_IN ? Ah[g] : (_Float16)Af[g];
        }
        // ---- stage B tile: 32x128, transposed into [n][k] ----
        #pragma unroll
        for (int i = 0; i < 16; ++i) {
            const int idx = tid + i * 256;
            const int kk = idx >> 7, n = idx & 127;
            const long g = (long)(k0 + kk) * N + (bn * BN + n);
            b_lds[n * BSTR + kk] = HALF_IN ? Bh[g] : (_Float16)Bf[g];
        }
        __syncthreads();

        // ---- A fragment (ISA 16-bit A 16x32 layout): per lane, rows M=lane%16,
        //      halves K=[kb..kb+7] then K=[kb+16..kb+23] ----
        const v8h alo = *(const v8h*)&a_lds[mrow * 32 + kb];
        const v8h ahi = *(const v8h*)&a_lds[mrow * 32 + kb + 16];
        const v16h afrag = __builtin_shufflevector(alo, ahi,
            0,1,2,3,4,5,6,7,8,9,10,11,12,13,14,15);

        #pragma unroll
        for (int ct = 0; ct < 8; ++ct) {
            // ---- B fragment (16-bit B 32x16): N=lane%16, K=[kkb..kkb+7] then [kkb+16..kkb+23] ----
            const int n = ct * 16 + (lane & 15);
            const v8h b0 = *(const v8h*)&b_lds[n * BSTR + kkb];
            const v8h b1 = *(const v8h*)&b_lds[n * BSTR + kkb + 16];
            const v16h bfrag = __builtin_shufflevector(b0, b1,
                0,1,2,3,4,5,6,7,8,9,10,11,12,13,14,15);
            acc[ct] = __builtin_amdgcn_wmma_f32_16x16x32_f16(
                false, afrag, false, bfrag, (short)0, acc[ct], false, false);
        }
        __syncthreads();
    }

    // ---- epilogue.  D layout: VGPR v, lane L -> M = v + 8*(L/16), N = L%16 ----
    const int rbase = bm * BM + wv * 16 + (lane >> 4) * 8;
    const int cbase = bn * BN + (lane & 15);

    if (FINAL) {
        float* outp = (float*)outv;
        const float cA2 = sc[0];
        const float cB  = sc[1];
        float wj[8];
        #pragma unroll
        for (int j = 0; j < 8; ++j) wj[j] = sc[2 + j];
        #pragma unroll
        for (int ct = 0; ct < 8; ++ct) {
            const int col = cbase + ct * 16;
            #pragma unroll
            for (int v = 0; v < 8; ++v) {
                const int r = rbase + v;
                const float* cp = coh + (long)r * 8192 + col;  // [B][P=8][H=1024]
                float s = 0.0f;
                #pragma unroll
                for (int j = 0; j < 8; ++j) s += wj[j] * cp[j * 1024];
                outp[(long)r * N + col] = cB * acc[ct][v] + cA2 * s;
            }
        }
    } else {
        _Float16* outp = (_Float16*)outv;
        #pragma unroll
        for (int ct = 0; ct < 8; ++ct) {
            const int col = cbase + ct * 16;
            #pragma unroll
            for (int v = 0; v < 8; ++v)
                outp[(long)(rbase + v) * N + col] = (_Float16)acc[ct][v];
        }
    }
}

// ---------------------------------------------------------------------------
// launch:  inputs: x[2048,1024], W_input[1024,1024], W_paths[8,1024,1024],
//          J[8,8], coherent_state[2048,8,1024], t[1]  (all f32)
// output:  [2048,1024] f32
// ws:      sc (64B @0) | W_eff f16 (2MB @256) | base f16 (4MB after)
// ---------------------------------------------------------------------------
extern "C" void kernel_launch(void* const* d_in, const int* in_sizes, int n_in,
                              void* d_out, int out_size, void* d_ws, size_t ws_size,
                              hipStream_t stream) {
    (void)in_sizes; (void)n_in; (void)out_size; (void)ws_size;
    const float* x   = (const float*)d_in[0];
    const float* Wi  = (const float*)d_in[1];
    const float* Wp  = (const float*)d_in[2];
    const float* J   = (const float*)d_in[3];
    const float* coh = (const float*)d_in[4];
    const float* t   = (const float*)d_in[5];
    float* out = (float*)d_out;

    char* ws = (char*)d_ws;
    float*    sc   = (float*)ws;                                   // 16 floats
    _Float16* weff = (_Float16*)(ws + 256);                        // 1024*1024 f16
    _Float16* base = (_Float16*)(ws + 256 + 2u * 1024u * 1024u);   // 2048*1024 f16

    qc_scalars_kernel<<<1, 64, 0, stream>>>(J, t, sc);
    qc_weff_kernel<<<4096, 256, 0, stream>>>(Wp, sc, weff);
    // base = x @ W_input   (f32 in, f16 staged, f16 out)
    qc_gemm_kernel<false, false><<<dim3(8, 16), 256, 0, stream>>>(
        x, Wi, base, nullptr, sc);
    // out = fused epilogue( base @ W_eff , coherent_state )
    qc_gemm_kernel<true, true><<<dim3(8, 16), 256, 0, stream>>>(
        base, weff, out, coh, sc);
}
